// SelfAttention_68032281968783
// MI455X (gfx1250) — matrix operand order repaired and checked by reading, not compiled, not run on hardware
//
#include <hip/hip_runtime.h>
#include <hip/hip_bf16.h>

// ---------------------------------------------------------------------------
// Self-attention (b=4, n=2048, dim=1024, heads=16, dim_head=64) for gfx1250.
// f16 WMMA everywhere (v_wmma_f32_16x16x32_f16, f32 accumulate), flash
// attention (scores never hit memory), V kept transposed in global so the
// attention staging is pure 2D copies done with GLOBAL_LOAD_ASYNC_TO_LDS_B128
// (ASYNCcnt) double-buffered against the WMMA stream. Softmax scale folded
// into Q at QKV-projection time.
// ---------------------------------------------------------------------------

typedef _Float16 f16;
typedef __attribute__((ext_vector_type(16))) _Float16 v16h;
typedef __attribute__((ext_vector_type(8)))  float    v8f;
typedef unsigned int u32x4 __attribute__((ext_vector_type(4)));
typedef u32x4 __attribute__((may_alias)) u32x4a;
typedef unsigned long long __attribute__((may_alias)) u64a;

union AFrag { v16h v; u32x4a q[2]; };
union H4   { f16 h[4]; unsigned long long u; };

static __device__ __forceinline__ v8f wmma16(v16h a, v16h b, v8f c) {
    // D = A(16x32 f16) * B(32x16 f16) + C(16x16 f32)
    return __builtin_amdgcn_wmma_f32_16x16x32_f16(
        /*neg_a=*/false, a, /*neg_b=*/false, b,
        /*c_mod=*/(short)0, c, /*reuse_a=*/false, /*reuse_b=*/false);
}

// Async 16B global->LDS copy (per lane), tracked by ASYNCcnt.
static __device__ __forceinline__ void async_cp16(unsigned lds_off,
                                                  const void* gptr) {
    unsigned long long ga = (unsigned long long)gptr;
    asm volatile("global_load_async_to_lds_b128 %0, %1, off"
                 :: "v"(lds_off), "v"(ga) : "memory");
}
static __device__ __forceinline__ void wait_async_le8() {
    asm volatile("s_wait_asynccnt 0x8" ::: "memory");
}
static __device__ __forceinline__ void wait_async_0() {
    asm volatile("s_wait_asynccnt 0x0" ::: "memory");
}

#define DIMK   1024
#define NQKV   3072
#define MROWS  8192   // 4*2048
#define HEADS  16
#define DHEAD  64
#define SEQ    2048

// ======================= Kernel 1: QKV projection ==========================
// C[8192,3072] = X[8192,1024](f32) @ W[1024,3072](f32) + b, scatter to
// Q/K f16 [B,H,N,64] and V f16 TRANSPOSED [B,H,64,N]; Q pre-scaled by 1/8.
#define T_M 128
#define T_N 128
#define T_K 32
#define LDT 40   // LDS row stride in halfs (32 + 8 pad), keeps 16B alignment

__global__ __launch_bounds__(256) void gemm_qkv_kernel(
    const float* __restrict__ X, const float* __restrict__ W,
    const float* __restrict__ bias, f16* __restrict__ Q,
    f16* __restrict__ K, f16* __restrict__ Vt)
{
    __shared__ f16 As[T_M * LDT];   // A tile, row-major [m][k]
    __shared__ f16 Bs[T_N * LDT];   // B tile, transposed  [n][k]

    const int tid  = threadIdx.x;
    const int wave = tid >> 5, lane = tid & 31;
    const int l16  = lane & 15, hi = lane >> 4;
    const int waveM = wave >> 2;          // 0..1  -> 64 rows each
    const int waveN = wave & 3;           // 0..3  -> 32 cols each
    const int row0 = blockIdx.x * T_M;
    const int col0 = blockIdx.y * T_N;

    v8f acc[4][2];
    #pragma unroll
    for (int i = 0; i < 4; ++i)
        #pragma unroll
        for (int j = 0; j < 2; ++j) acc[i][j] = (v8f)0.0f;

    // register-stage pipeline: tile kt+1 global loads issued during tile kt
    float4 pa[4], pb[4];
    #pragma unroll
    for (int i = 0; i < 4; ++i) {
        int idx = tid + i * 256;
        int r = idx >> 3, c4 = idx & 7;
        pa[i] = *(const float4*)(X + (size_t)(row0 + r) * DIMK + c4 * 4);
        int kk = idx >> 5, c4b = idx & 31;
        pb[i] = *(const float4*)(W + (size_t)kk * NQKV + col0 + c4b * 4);
    }

    for (int kt = 0; kt < DIMK / T_K; ++kt) {
        __syncthreads();
        // ---- fill LDS from staged registers (convert to f16; B transposed)
        #pragma unroll
        for (int i = 0; i < 4; ++i) {
            int idx = tid + i * 256;
            int r = idx >> 3, c4 = idx & 7;
            H4 h; h.h[0] = (f16)pa[i].x; h.h[1] = (f16)pa[i].y;
                  h.h[2] = (f16)pa[i].z; h.h[3] = (f16)pa[i].w;
            *(u64a*)(&As[r * LDT + c4 * 4]) = h.u;
            int kk = idx >> 5, c4b = idx & 31;
            Bs[(c4b * 4 + 0) * LDT + kk] = (f16)pb[i].x;
            Bs[(c4b * 4 + 1) * LDT + kk] = (f16)pb[i].y;
            Bs[(c4b * 4 + 2) * LDT + kk] = (f16)pb[i].z;
            Bs[(c4b * 4 + 3) * LDT + kk] = (f16)pb[i].w;
        }
        // ---- issue next tile's global loads (in flight across compute)
        if (kt + 1 < DIMK / T_K) {
            #pragma unroll
            for (int i = 0; i < 4; ++i) {
                int idx = tid + i * 256;
                int r = idx >> 3, c4 = idx & 7;
                pa[i] = *(const float4*)(X + (size_t)(row0 + r) * DIMK
                                           + (kt + 1) * T_K + c4 * 4);
                int kk = idx >> 5, c4b = idx & 31;
                pb[i] = *(const float4*)(W + (size_t)((kt + 1) * T_K + kk) * NQKV
                                           + col0 + c4b * 4);
            }
        }
        __syncthreads();

        // ---- fragments: A rows interleave K 0-7/16-23 (lanes 0-15) vs
        //      8-15/24-31 (lanes 16-31); B cols use the same interleave.
        AFrag af[4];
        #pragma unroll
        for (int mi = 0; mi < 4; ++mi) {
            int base = (waveM * 64 + mi * 16 + l16) * LDT + hi * 8;
            af[mi].q[0] = *(const u32x4a*)(&As[base]);
            af[mi].q[1] = *(const u32x4a*)(&As[base + 16]);
        }
        AFrag bf[2];
        #pragma unroll
        for (int ni = 0; ni < 2; ++ni) {
            int base = (waveN * 32 + ni * 16 + l16) * LDT + hi * 8;
            bf[ni].q[0] = *(const u32x4a*)(&Bs[base]);
            bf[ni].q[1] = *(const u32x4a*)(&Bs[base + 16]);
        }
        #pragma unroll
        for (int mi = 0; mi < 4; ++mi)
            #pragma unroll
            for (int ni = 0; ni < 2; ++ni)
                acc[mi][ni] = wmma16(af[mi].v, bf[ni].v, acc[mi][ni]);
    }

    // ---- epilogue: +bias; Q,K -> [B,H,N,64], V -> [B,H,64,N]; q *= 1/8
    #pragma unroll
    for (int ni = 0; ni < 2; ++ni) {
        int col = col0 + waveN * 32 + ni * 16 + l16;    // < 3072
        float bv = bias[col];
        int which = col >> 10;                           // 0=q 1=k 2=v
        int rem = col & 1023, hh = rem >> 6, dd = rem & 63;
        f16* dst = (which == 0) ? Q : (which == 1) ? K : Vt;
        float sc = (which == 0) ? 0.125f : 1.0f;
        #pragma unroll
        for (int mi = 0; mi < 4; ++mi) {
            #pragma unroll
            for (int v = 0; v < 8; ++v) {
                int row = row0 + waveM * 64 + mi * 16 + hi * 8 + v;  // < 8192
                int bb = row >> 11, nn = row & 2047;
                float val = (acc[mi][ni][v] + bv) * sc;
                size_t off;
                if (which == 2)
                    off = (((size_t)(bb * HEADS + hh)) * DHEAD + dd) * SEQ + nn;
                else
                    off = (((size_t)(bb * HEADS + hh)) * SEQ + nn) * DHEAD + dd;
                dst[off] = (f16)val;
            }
        }
    }
}

// ====================== Kernel 2: flash attention ==========================
// One workgroup per (b, h, 64 q-rows). 4 waves, 16 q-rows each. K/V tiles
// staged by double-buffered GLOBAL_LOAD_ASYNC_TO_LDS_B128 copies (V already
// transposed in global). S=Q*K^T, online softmax via shfl_xor reductions
// matched to the C/D VGPR layout, P re-swizzled through per-wave LDS,
// O += P*V. Output [B,N,H*64] f16.
#define LDK 72   // 64 + 8 pad halfs
#define NTILES (SEQ / 64)

__global__ __launch_bounds__(128) void attn_kernel(
    const f16* __restrict__ Q, const f16* __restrict__ K,
    const f16* __restrict__ Vt, f16* __restrict__ O)
{
    __shared__ f16 Ksb[2][64 * LDK];    // K tile, row-major [m][d], x2 buffers
    __shared__ f16 Vtb[2][64 * LDK];    // V tile, [d][m], x2 buffers
    __shared__ f16 Pw[4 * 16 * LDK];    // per-wave P buffer [r][m]

    const int tid  = threadIdx.x;
    const int wave = tid >> 5, lane = tid & 31;
    const int l16  = lane & 15, hi = lane >> 4;
    const int qt = blockIdx.x, h = blockIdx.y, b = blockIdx.z;

    const size_t headOff = ((size_t)(b * HEADS + h)) * SEQ * DHEAD;
    const f16* Qg  = Q  + headOff + (size_t)(qt * 64) * DHEAD;
    const f16* Kg  = K  + headOff;
    const f16* Vg  = Vt + headOff;      // [64][SEQ]

    // this thread's 8 copy slots per tile (4 K chunks + 4 V chunks of 16B)
    int cr[4], cc[4];
    #pragma unroll
    for (int i = 0; i < 4; ++i) {
        int idx = tid + i * 128;        // 0..511
        cr[i] = idx >> 3;               // row 0..63
        cc[i] = idx & 7;                // 16B chunk 0..7
    }

    // Q rows of this wave held entirely in registers (2 A-fragments, K=64)
    const int qrow = wave * 16 + l16;
    AFrag qf[2];
    #pragma unroll
    for (int s = 0; s < 2; ++s) {
        qf[s].q[0] = *(const u32x4a*)(Qg + qrow * DHEAD + s * 32 + hi * 8);
        qf[s].q[1] = *(const u32x4a*)(Qg + qrow * DHEAD + s * 32 + 16 + hi * 8);
    }

    float mrun[8], lrun[8];
    v8f oacc[4];
    #pragma unroll
    for (int v = 0; v < 8; ++v) { mrun[v] = -3.0e38f; lrun[v] = 0.0f; }
    #pragma unroll
    for (int nd = 0; nd < 4; ++nd) oacc[nd] = (v8f)0.0f;

    f16* myP = &Pw[wave * 16 * LDK];

    // prologue: async-copy tile 0 into buffer 0 (8 async insts per wave)
    #pragma unroll
    for (int i = 0; i < 4; ++i) {
        async_cp16((unsigned)(size_t)&Ksb[0][cr[i] * LDK + cc[i] * 8],
                   Kg + (size_t)cr[i] * DHEAD + cc[i] * 8);
        async_cp16((unsigned)(size_t)&Vtb[0][cr[i] * LDK + cc[i] * 8],
                   Vg + (size_t)cr[i] * SEQ + cc[i] * 8);
    }

    for (int t = 0; t < NTILES; ++t) {
        const int cur = t & 1;
        // issue tile t+1 into the other buffer (freed by last iter's barrier)
        if (t + 1 < NTILES) {
            const f16* Kn = Kg + (size_t)((t + 1) * 64) * DHEAD;
            const f16* Vn = Vg + (size_t)((t + 1) * 64);
            #pragma unroll
            for (int i = 0; i < 4; ++i) {
                async_cp16((unsigned)(size_t)&Ksb[cur ^ 1][cr[i] * LDK + cc[i] * 8],
                           Kn + (size_t)cr[i] * DHEAD + cc[i] * 8);
                async_cp16((unsigned)(size_t)&Vtb[cur ^ 1][cr[i] * LDK + cc[i] * 8],
                           Vn + (size_t)cr[i] * SEQ + cc[i] * 8);
            }
            wait_async_le8();   // in-order: <=8 outstanding => tile t landed
        } else {
            wait_async_0();
        }
        __syncthreads();        // everyone's portion of tile t is in LDS

        const f16* Ks = Ksb[cur];
        const f16* Vs = Vtb[cur];

        // ---- S = Q * K^T  (Q pre-scaled by 1/8)
        v8f sacc[4];
        #pragma unroll
        for (int ni = 0; ni < 4; ++ni) sacc[ni] = (v8f)0.0f;
        #pragma unroll
        for (int ni = 0; ni < 4; ++ni) {
            #pragma unroll
            for (int s = 0; s < 2; ++s) {
                AFrag kf;
                int base = (ni * 16 + l16) * LDK + s * 32 + hi * 8;
                kf.q[0] = *(const u32x4a*)(&Ks[base]);
                kf.q[1] = *(const u32x4a*)(&Ks[base + 16]);
                sacc[ni] = wmma16(qf[s].v, kf.v, sacc[ni]);
            }
        }

        // ---- online softmax; D layout: lane = column, VGPR v = row hi*8+v
        #pragma unroll
        for (int v = 0; v < 8; ++v) {
            float mx = sacc[0][v];
            #pragma unroll
            for (int ni = 1; ni < 4; ++ni) mx = fmaxf(mx, sacc[ni][v]);
            #pragma unroll
            for (int d = 1; d < 16; d <<= 1) mx = fmaxf(mx, __shfl_xor(mx, d, 32));
            float mnew = fmaxf(mrun[v], mx);
            float alpha = __expf(mrun[v] - mnew);
            mrun[v] = mnew;
            float rs = 0.0f;
            #pragma unroll
            for (int ni = 0; ni < 4; ++ni) {
                float p = __expf(sacc[ni][v] - mnew);
                rs += p;
                myP[(hi * 8 + v) * LDK + ni * 16 + l16] = (f16)p;
            }
            #pragma unroll
            for (int d = 1; d < 16; d <<= 1) rs += __shfl_xor(rs, d, 32);
            lrun[v] = lrun[v] * alpha + rs;
            #pragma unroll
            for (int nd = 0; nd < 4; ++nd) oacc[nd][v] *= alpha;
        }

        // ---- O += P * V  (same-wave LDS round trip for P; in-order per wave)
        #pragma unroll
        for (int s2 = 0; s2 < 2; ++s2) {
            AFrag pf;
            int pbase = l16 * LDK + s2 * 32 + hi * 8;
            pf.q[0] = *(const u32x4a*)(&myP[pbase]);
            pf.q[1] = *(const u32x4a*)(&myP[pbase + 16]);
            #pragma unroll
            for (int nd = 0; nd < 4; ++nd) {
                AFrag vf;
                int vbase = (nd * 16 + l16) * LDK + s2 * 32 + hi * 8;
                vf.q[0] = *(const u32x4a*)(&Vs[vbase]);
                vf.q[1] = *(const u32x4a*)(&Vs[vbase + 16]);
                oacc[nd] = wmma16(pf.v, vf.v, oacc[nd]);
            }
        }
        __syncthreads();        // tile t fully consumed; buffer reusable
    }

    // ---- finalize: O /= l, store [B, N, H*64] f16 for the output GEMM
    #pragma unroll
    for (int nd = 0; nd < 4; ++nd) {
        #pragma unroll
        for (int v = 0; v < 8; ++v) {
            int r = hi * 8 + v;
            float val = oacc[nd][v] / lrun[v];
            size_t off = ((size_t)(b * SEQ + qt * 64 + wave * 16 + r)) * 1024
                       + h * DHEAD + nd * 16 + l16;
            O[off] = (f16)val;
        }
    }
}

// ====================== Kernel 3: output projection ========================
// out[8192,1024](f32) = O[8192,1024](f16) @ w_out[1024,1024](f32->f16) + b
__global__ __launch_bounds__(256) void gemm_out_kernel(
    const f16* __restrict__ A, const float* __restrict__ W,
    const float* __restrict__ bias, float* __restrict__ out)
{
    __shared__ f16 As[T_M * LDT];
    __shared__ f16 Bs[T_N * LDT];

    const int tid  = threadIdx.x;
    const int wave = tid >> 5, lane = tid & 31;
    const int l16  = lane & 15, hi = lane >> 4;
    const int waveM = wave >> 2, waveN = wave & 3;
    const int row0 = blockIdx.x * T_M;
    const int col0 = blockIdx.y * T_N;

    v8f acc[4][2];
    #pragma unroll
    for (int i = 0; i < 4; ++i)
        #pragma unroll
        for (int j = 0; j < 2; ++j) acc[i][j] = (v8f)0.0f;

    u32x4 pa[2]; float4 pb[4];
    #pragma unroll
    for (int i = 0; i < 2; ++i) {
        int idx = tid + i * 256;
        int r = idx >> 2, c = idx & 3;
        pa[i] = *(const u32x4a*)(A + (size_t)(row0 + r) * DIMK + c * 8);
    }
    #pragma unroll
    for (int i = 0; i < 4; ++i) {
        int idx = tid + i * 256;
        int kk = idx >> 5, c4 = idx & 31;
        pb[i] = *(const float4*)(W + (size_t)kk * 1024 + col0 + c4 * 4);
    }

    for (int kt = 0; kt < DIMK / T_K; ++kt) {
        __syncthreads();
        #pragma unroll
        for (int i = 0; i < 2; ++i) {
            int idx = tid + i * 256;
            int r = idx >> 2, c = idx & 3;
            *(u32x4a*)(&As[r * LDT + c * 8]) = pa[i];
        }
        #pragma unroll
        for (int i = 0; i < 4; ++i) {
            int idx = tid + i * 256;
            int kk = idx >> 5, c4 = idx & 31;
            Bs[(c4 * 4 + 0) * LDT + kk] = (f16)pb[i].x;
            Bs[(c4 * 4 + 1) * LDT + kk] = (f16)pb[i].y;
            Bs[(c4 * 4 + 2) * LDT + kk] = (f16)pb[i].z;
            Bs[(c4 * 4 + 3) * LDT + kk] = (f16)pb[i].w;
        }
        if (kt + 1 < DIMK / T_K) {
            #pragma unroll
            for (int i = 0; i < 2; ++i) {
                int idx = tid + i * 256;
                int r = idx >> 2, c = idx & 3;
                pa[i] = *(const u32x4a*)(A + (size_t)(row0 + r) * DIMK
                                           + (kt + 1) * T_K + c * 8);
            }
            #pragma unroll
            for (int i = 0; i < 4; ++i) {
                int idx = tid + i * 256;
                int kk = idx >> 5, c4 = idx & 31;
                pb[i] = *(const float4*)(W + (size_t)((kt + 1) * T_K + kk) * 1024
                                           + col0 + c4 * 4);
            }
        }
        __syncthreads();

        AFrag af[4];
        #pragma unroll
        for (int mi = 0; mi < 4; ++mi) {
            int base = (waveM * 64 + mi * 16 + l16) * LDT + hi * 8;
            af[mi].q[0] = *(const u32x4a*)(&As[base]);
            af[mi].q[1] = *(const u32x4a*)(&As[base + 16]);
        }
        AFrag bf[2];
        #pragma unroll
        for (int ni = 0; ni < 2; ++ni) {
            int base = (waveN * 32 + ni * 16 + l16) * LDT + hi * 8;
            bf[ni].q[0] = *(const u32x4a*)(&Bs[base]);
            bf[ni].q[1] = *(const u32x4a*)(&Bs[base + 16]);
        }
        #pragma unroll
        for (int mi = 0; mi < 4; ++mi)
            #pragma unroll
            for (int ni = 0; ni < 2; ++ni)
                acc[mi][ni] = wmma16(af[mi].v, bf[ni].v, acc[mi][ni]);
    }

    #pragma unroll
    for (int ni = 0; ni < 2; ++ni) {
        int col = col0 + waveN * 32 + ni * 16 + l16;
        float bv = bias[col];
        #pragma unroll
        for (int mi = 0; mi < 4; ++mi) {
            #pragma unroll
            for (int v = 0; v < 8; ++v) {
                int row = row0 + waveM * 64 + mi * 16 + hi * 8 + v;
                out[(size_t)row * 1024 + col] = acc[mi][ni][v] + bv;
            }
        }
    }
}

// ============================ launch ======================================
extern "C" void kernel_launch(void* const* d_in, const int* in_sizes, int n_in,
                              void* d_out, int out_size, void* d_ws, size_t ws_size,
                              hipStream_t stream) {
    const float* x     = (const float*)d_in[0];   // [4,2048,1024]
    const float* w_qkv = (const float*)d_in[1];   // [1024,3072]
    const float* b_qkv = (const float*)d_in[2];   // [3072]
    const float* w_out = (const float*)d_in[3];   // [1024,1024]
    const float* b_out = (const float*)d_in[4];   // [1024]
    float* out = (float*)d_out;

    // workspace: Q,K [B,H,N,64], V [B,H,64,N], O [B,N,1024] f16 = 4 * 16 MiB
    char* ws = (char*)d_ws;
    const size_t seg = (size_t)MROWS * 1024 * sizeof(f16);   // 16 MiB
    f16* Qh  = (f16*)(ws);
    f16* Kh  = (f16*)(ws + seg);
    f16* Vth = (f16*)(ws + 2 * seg);
    f16* Oh  = (f16*)(ws + 3 * seg);

    gemm_qkv_kernel<<<dim3(MROWS / T_M, NQKV / T_N), 256, 0, stream>>>(
        x, w_qkv, b_qkv, Qh, Kh, Vth);
    attn_kernel<<<dim3(SEQ / 64, HEADS, 4), 128, 0, stream>>>(Qh, Kh, Vth, Oh);
    gemm_out_kernel<<<dim3(MROWS / T_M, 1024 / T_N), 256, 0, stream>>>(
        Oh, w_out, b_out, out);
}
